// TransformerBlock_71262097375467
// MI455X (gfx1250) — matrix operand order repaired and checked by reading, not compiled, not run on hardware
//
#include <hip/hip_runtime.h>

// ---------------------------------------------------------------------------
// Transformer block for MI455X (gfx1250, wave32, WMMA bf16 16x16x32).
// B=2, S=2048, D=768, H=12, DEPTH=64, DFF=3072. All GEMMs run on
// v_wmma_f32_16x16x32_bf16 with fp32 accumulation; softmax/LN in fp32.
// GEMM: 128x64 block tile, 32x64 wave tile (8 WMMA / k-step, 1.5 ds_load_b128
// per WMMA), double-buffered LDS, fp32->bf16 conversion at staging.
// ---------------------------------------------------------------------------

typedef unsigned short ushort_t;
typedef __bf16 bf16x16 __attribute__((ext_vector_type(16)));
typedef float f32x8 __attribute__((ext_vector_type(8)));

#define HND 12
#define SEQ 2048
#define DM 768
#define LDT 40 // GEMM LDS row stride in halves (32 data + 8 pad = 80B)
#define LQ 72  // attn LDS row stride in halves (64 data + 8 pad = 144B)

struct alignas(16) U128 {
  unsigned a, b, c, d;
};
union FragU {
  U128 q[2];
  bf16x16 v;
};

__device__ __forceinline__ unsigned pack_bf2(float x, float y) {
  union {
    float f;
    unsigned u;
  } a, b;
  a.f = x;
  b.f = y;
  unsigned lo = (a.u + 0x7FFFu + ((a.u >> 16) & 1u)) >> 16;
  unsigned hi = (b.u + 0x7FFFu + ((b.u >> 16) & 1u)) >> 16;
  return lo | (hi << 16);
}
__device__ __forceinline__ ushort_t bf1(float x) {
  union {
    float f;
    unsigned u;
  } a;
  a.f = x;
  return (ushort_t)((a.u + 0x7FFFu + ((a.u >> 16) & 1u)) >> 16);
}

// A fragment (16x32 bf16): lane<16 row=lane, K0..7 & K16..23; lane>=16 same
// row, K8..15 & K24..31  ->  two ds_load_b128 at rowbase+{0,16}B and +32B.
__device__ __forceinline__ bf16x16 ld_frag_a(const ushort_t* p, int stride) {
  int lane = threadIdx.x & 31;
  int off = (lane & 15) * stride + ((lane >> 4) << 3); // halves
  FragU u;
  u.q[0] = *(const U128*)(p + off);
  u.q[1] = *(const U128*)(p + off + 16);
  return u.v;
}
// B fragment (32x16 bf16) from transposed LDS tile BT[n][k]:
// lane<16 col n=lane holds K0..7 & K16..23; lane>=16 holds K8..15 & K24..31.
__device__ __forceinline__ bf16x16 ld_frag_b(const ushort_t* p, int stride) {
  int lane = threadIdx.x & 31;
  int off = (lane & 15) * stride + ((lane >> 4) << 3); // halves
  FragU u;
  u.q[0] = *(const U128*)(p + off);
  u.q[1] = *(const U128*)(p + off + 16);
  return u.v;
}

// ---------------------------------------------------------------------------
// Generic GEMM: C[M,N] = A[M,K] * B[K,N]  (fp32 in, bf16 WMMA, fp32 out)
// Block tile 128(M) x 64(N), 4 waves, each wave 32x64.
// EPI==0: epilogue = (+bias)(+res)(relu) -> fp32 C
// EPI==1: QKV split epilogue -> bf16 q (pre-scaled 1/8), k, v-transposed
// ---------------------------------------------------------------------------
template <int EPI>
__global__ void __launch_bounds__(128)
    gemm_kernel(const float* __restrict__ A, const float* __restrict__ B,
                const float* __restrict__ bias, const float* __restrict__ res,
                float* __restrict__ C, ushort_t* __restrict__ qb,
                ushort_t* __restrict__ kb, ushort_t* __restrict__ vtb, int M,
                int N, int K, int has_bias, int has_res, int do_relu) {
  __shared__ __align__(16) ushort_t sa[2][128 * LDT];
  __shared__ __align__(16) ushort_t sb[2][64 * LDT];
  const int t = threadIdx.x;
  const int wave = t >> 5;
  const int lane = t & 31;
  const int m0 = blockIdx.y * 128;
  const int n0 = blockIdx.x * 64;
  const int nk = K >> 5;

  float4 ra[8], rb[4];
  // prologue: load k-step 0 into regs, store to LDS buffer 0
#pragma unroll
  for (int i = 0; i < 8; ++i) {
    int id = t + 128 * i;
    int row = id >> 3, kk = (id & 7) << 2;
    ra[i] = *(const float4*)(A + (size_t)(m0 + row) * K + kk);
  }
#pragma unroll
  for (int i = 0; i < 4; ++i) {
    int id = t + 128 * i;
    int kr = id >> 4, nn = (id & 15) << 2;
    rb[i] = *(const float4*)(B + (size_t)kr * N + n0 + nn);
  }
#pragma unroll
  for (int i = 0; i < 8; ++i) {
    int id = t + 128 * i;
    int row = id >> 3, kk = (id & 7) << 2;
    *(unsigned*)(&sa[0][row * LDT + kk]) = pack_bf2(ra[i].x, ra[i].y);
    *(unsigned*)(&sa[0][row * LDT + kk + 2]) = pack_bf2(ra[i].z, ra[i].w);
  }
#pragma unroll
  for (int i = 0; i < 4; ++i) {
    int id = t + 128 * i;
    int kr = id >> 4, nn = (id & 15) << 2;
    sb[0][(nn + 0) * LDT + kr] = bf1(rb[i].x);
    sb[0][(nn + 1) * LDT + kr] = bf1(rb[i].y);
    sb[0][(nn + 2) * LDT + kr] = bf1(rb[i].z);
    sb[0][(nn + 3) * LDT + kr] = bf1(rb[i].w);
  }
  __syncthreads();

  f32x8 acc[2][4];
#pragma unroll
  for (int mi = 0; mi < 2; ++mi)
#pragma unroll
    for (int j = 0; j < 4; ++j)
#pragma unroll
      for (int r = 0; r < 8; ++r) acc[mi][j][r] = 0.0f;

  for (int ks = 0; ks < nk; ++ks) {
    const int cur = ks & 1;
    const bool more = (ks + 1 < nk);
    if (more) {
      const int kbase = (ks + 1) << 5;
#pragma unroll
      for (int i = 0; i < 8; ++i) {
        int id = t + 128 * i;
        int row = id >> 3, kk = (id & 7) << 2;
        ra[i] = *(const float4*)(A + (size_t)(m0 + row) * K + kbase + kk);
      }
#pragma unroll
      for (int i = 0; i < 4; ++i) {
        int id = t + 128 * i;
        int kr = id >> 4, nn = (id & 15) << 2;
        rb[i] = *(const float4*)(B + (size_t)(kbase + kr) * N + n0 + nn);
      }
    }
    // compute current k-step: 2 A-frags + 4 B-frags -> 8 WMMAs per wave
    bf16x16 af0 = ld_frag_a(&sa[cur][(wave * 32) * LDT], LDT);
    bf16x16 af1 = ld_frag_a(&sa[cur][(wave * 32 + 16) * LDT], LDT);
#pragma unroll
    for (int j = 0; j < 4; ++j) {
      bf16x16 bfr = ld_frag_b(&sb[cur][j * 16 * LDT], LDT);
      acc[0][j] = __builtin_amdgcn_wmma_f32_16x16x32_bf16(
          false, af0, false, bfr, (short)0, acc[0][j], false, false);
      acc[1][j] = __builtin_amdgcn_wmma_f32_16x16x32_bf16(
          false, af1, false, bfr, (short)0, acc[1][j], false, false);
    }
    if (more) {
      const int nxt = cur ^ 1;
#pragma unroll
      for (int i = 0; i < 8; ++i) {
        int id = t + 128 * i;
        int row = id >> 3, kk = (id & 7) << 2;
        *(unsigned*)(&sa[nxt][row * LDT + kk]) = pack_bf2(ra[i].x, ra[i].y);
        *(unsigned*)(&sa[nxt][row * LDT + kk + 2]) = pack_bf2(ra[i].z, ra[i].w);
      }
#pragma unroll
      for (int i = 0; i < 4; ++i) {
        int id = t + 128 * i;
        int kr = id >> 4, nn = (id & 15) << 2;
        sb[nxt][(nn + 0) * LDT + kr] = bf1(rb[i].x);
        sb[nxt][(nn + 1) * LDT + kr] = bf1(rb[i].y);
        sb[nxt][(nn + 2) * LDT + kr] = bf1(rb[i].z);
        sb[nxt][(nn + 3) * LDT + kr] = bf1(rb[i].w);
      }
    }
    __syncthreads();
  }

  // Epilogue.  C layout: VGPR r, lanes 0-15 -> (M=r, N=lane); lanes 16-31 ->
  // (M=r+8, N=lane-16).
  if (EPI == 0) {
#pragma unroll
    for (int mi = 0; mi < 2; ++mi) {
#pragma unroll
      for (int j = 0; j < 4; ++j) {
        const int col = n0 + j * 16 + (lane & 15);
        const float bv = has_bias ? bias[col] : 0.0f;
#pragma unroll
        for (int r = 0; r < 8; ++r) {
          const int row = m0 + wave * 32 + mi * 16 + r + ((lane >> 4) << 3);
          float v = acc[mi][j][r] + bv;
          if (has_res) v += res[(size_t)row * N + col];
          if (do_relu) v = fmaxf(v, 0.0f);
          C[(size_t)row * N + col] = v;
        }
      }
    }
  } else {
#pragma unroll
    for (int mi = 0; mi < 2; ++mi) {
#pragma unroll
      for (int j = 0; j < 4; ++j) {
        const int colg = n0 + j * 16 + (lane & 15);
        const int part = colg / DM; // 0=q 1=k 2=v (tile never straddles parts)
        const int c = colg - part * DM;
        const int hh = c >> 6;
        const int dd = c & 63;
#pragma unroll
        for (int r = 0; r < 8; ++r) {
          const int row = m0 + wave * 32 + mi * 16 + r + ((lane >> 4) << 3);
          const int bb = row >> 11; // /S
          const int ss = row & (SEQ - 1);
          const float v = acc[mi][j][r];
          if (part == 0) {
            qb[((size_t)(bb * HND + hh) * SEQ + ss) * 64 + dd] =
                bf1(v * 0.125f);
          } else if (part == 1) {
            kb[((size_t)(bb * HND + hh) * SEQ + ss) * 64 + dd] = bf1(v);
          } else {
            vtb[((size_t)(bb * HND + hh) * 64 + dd) * SEQ + ss] = bf1(v);
          }
        }
      }
    }
  }
}

// ---------------------------------------------------------------------------
// Flash attention: one WG (4 waves) per (b*h, 64-row q-tile); online softmax.
// ---------------------------------------------------------------------------
__global__ void __launch_bounds__(128)
    attn_kernel(const ushort_t* __restrict__ qb, const ushort_t* __restrict__ kb,
                const ushort_t* __restrict__ vtb, float* __restrict__ ctx) {
  __shared__ __align__(16) ushort_t sq[64 * LQ];
  __shared__ __align__(16) ushort_t sk[64 * LQ];
  __shared__ __align__(16) ushort_t sv[64 * LQ];
  __shared__ __align__(16) ushort_t sp[64 * LQ];
  const int t = threadIdx.x;
  const int wave = t >> 5;
  const int lane = t & 31;
  const int qt = blockIdx.x;
  const int bh = blockIdx.y;

  // stage Q tile (bf16 copy, rows padded to LQ halves)
#pragma unroll
  for (int i = 0; i < 4; ++i) {
    int id = t + 128 * i;
    int row = id >> 3, c = id & 7;
    const U128* src =
        (const U128*)(qb + ((size_t)bh * SEQ + qt * 64 + row) * 64);
    *(U128*)(&sq[row * LQ + c * 8]) = src[c];
  }

  f32x8 o[4];
#pragma unroll
  for (int j = 0; j < 4; ++j)
#pragma unroll
    for (int r = 0; r < 8; ++r) o[j][r] = 0.0f;
  float mprev[8], lsum[8];
#pragma unroll
  for (int r = 0; r < 8; ++r) {
    mprev[r] = -1e30f;
    lsum[r] = 0.0f;
  }

  for (int jt = 0; jt <= qt; ++jt) {
    __syncthreads(); // Q staged (iter 0); sk/sv free from previous iter
#pragma unroll
    for (int i = 0; i < 4; ++i) {
      int id = t + 128 * i;
      int row = id >> 3, c = id & 7;
      const U128* ksrc =
          (const U128*)(kb + ((size_t)bh * SEQ + jt * 64 + row) * 64);
      *(U128*)(&sk[row * LQ + c * 8]) = ksrc[c];
      const U128* vsrc =
          (const U128*)(vtb + ((size_t)bh * 64 + row) * SEQ + jt * 64);
      *(U128*)(&sv[row * LQ + c * 8]) = vsrc[c];
    }
    __syncthreads();

    // scores tile: wave owns 16 q rows x 64 k cols
    f32x8 sc[4];
#pragma unroll
    for (int j = 0; j < 4; ++j)
#pragma unroll
      for (int r = 0; r < 8; ++r) sc[j][r] = 0.0f;
#pragma unroll
    for (int kk = 0; kk < 2; ++kk) {
      bf16x16 af = ld_frag_a(sq + wave * 16 * LQ + kk * 32, LQ);
#pragma unroll
      for (int j = 0; j < 4; ++j) {
        bf16x16 bfr = ld_frag_b(sk + j * 16 * LQ + kk * 32, LQ);
        sc[j] = __builtin_amdgcn_wmma_f32_16x16x32_bf16(
            false, af, false, bfr, (short)0, sc[j], false, false);
      }
    }
    if (jt == qt) { // causal mask on diagonal tile
#pragma unroll
      for (int j = 0; j < 4; ++j) {
        const int cc = j * 16 + (lane & 15);
#pragma unroll
        for (int r = 0; r < 8; ++r) {
          const int rr = wave * 16 + r + ((lane >> 4) << 3);
          if (cc > rr) sc[j][r] = -1e30f;
        }
      }
    }
    // online softmax (row lives in one 16-lane group -> xor 1/2/4/8)
    float alpha[8];
#pragma unroll
    for (int r = 0; r < 8; ++r) {
      float m = fmaxf(fmaxf(sc[0][r], sc[1][r]), fmaxf(sc[2][r], sc[3][r]));
      m = fmaxf(m, __shfl_xor(m, 1));
      m = fmaxf(m, __shfl_xor(m, 2));
      m = fmaxf(m, __shfl_xor(m, 4));
      m = fmaxf(m, __shfl_xor(m, 8));
      const float mi = fmaxf(mprev[r], m);
      alpha[r] = __expf(mprev[r] - mi);
      float rs = 0.0f;
#pragma unroll
      for (int j = 0; j < 4; ++j) {
        const float p = __expf(sc[j][r] - mi);
        sc[j][r] = p;
        rs += p;
      }
      rs += __shfl_xor(rs, 1);
      rs += __shfl_xor(rs, 2);
      rs += __shfl_xor(rs, 4);
      rs += __shfl_xor(rs, 8);
      lsum[r] = lsum[r] * alpha[r] + rs;
      mprev[r] = mi;
    }
#pragma unroll
    for (int j = 0; j < 4; ++j)
#pragma unroll
      for (int r = 0; r < 8; ++r) o[j][r] *= alpha[r];

    // P (bf16) -> LDS rows owned by this wave; in-wave DS ordering makes the
    // following A-fragment reads safe without a barrier.
#pragma unroll
    for (int j = 0; j < 4; ++j) {
      const int cc = j * 16 + (lane & 15);
#pragma unroll
      for (int r = 0; r < 8; ++r) {
        const int rr = wave * 16 + r + ((lane >> 4) << 3);
        sp[rr * LQ + cc] = bf1(sc[j][r]);
      }
    }
    // O += P @ V  (Vt tile is BT[n=d][k=s_k])
#pragma unroll
    for (int kk = 0; kk < 2; ++kk) {
      bf16x16 af = ld_frag_a(sp + wave * 16 * LQ + kk * 32, LQ);
#pragma unroll
      for (int j = 0; j < 4; ++j) {
        bf16x16 bfr = ld_frag_b(sv + j * 16 * LQ + kk * 32, LQ);
        o[j] = __builtin_amdgcn_wmma_f32_16x16x32_bf16(
            false, af, false, bfr, (short)0, o[j], false, false);
      }
    }
  }

  // write ctx merged as [b][s][h*64+d] fp32
  const int bb = bh / HND;
  const int hh = bh % HND;
#pragma unroll
  for (int j = 0; j < 4; ++j) {
    const int dd = j * 16 + (lane & 15);
#pragma unroll
    for (int r = 0; r < 8; ++r) {
      const int ss = qt * 64 + wave * 16 + r + ((lane >> 4) << 3);
      ctx[((size_t)bb * SEQ + ss) * DM + hh * 64 + dd] = o[j][r] / lsum[r];
    }
  }
}

// ---------------------------------------------------------------------------
// LayerNorm over D=768, one 256-thread block per row.
// ---------------------------------------------------------------------------
__global__ void __launch_bounds__(256)
    ln_kernel(const float* __restrict__ x, const float* __restrict__ g,
              const float* __restrict__ be, float* __restrict__ out) {
  __shared__ float s1[256];
  __shared__ float s2[256];
  const int row = blockIdx.x;
  const float* xr = x + (size_t)row * DM;
  float v[3], a = 0.0f, b = 0.0f;
#pragma unroll
  for (int i = 0; i < 3; ++i) {
    v[i] = xr[threadIdx.x + 256 * i];
    a += v[i];
    b += v[i] * v[i];
  }
  s1[threadIdx.x] = a;
  s2[threadIdx.x] = b;
  __syncthreads();
  for (int off = 128; off > 0; off >>= 1) {
    if (threadIdx.x < off) {
      s1[threadIdx.x] += s1[threadIdx.x + off];
      s2[threadIdx.x] += s2[threadIdx.x + off];
    }
    __syncthreads();
  }
  const float mean = s1[0] * (1.0f / DM);
  const float var = s2[0] * (1.0f / DM) - mean * mean;
  const float rstd = rsqrtf(var + 1e-5f);
#pragma unroll
  for (int i = 0; i < 3; ++i) {
    const int c = threadIdx.x + 256 * i;
    out[(size_t)row * DM + c] = g[c] * (v[i] - mean) * rstd + be[c];
  }
}

// ---------------------------------------------------------------------------
extern "C" void kernel_launch(void* const* d_in, const int* in_sizes, int n_in,
                              void* d_out, int out_size, void* d_ws,
                              size_t ws_size, hipStream_t stream) {
  const float* x = (const float*)d_in[0];
  const float* Wqkv = (const float*)d_in[1];
  const float* Wout = (const float*)d_in[2];
  const float* bout = (const float*)d_in[3];
  const float* W1 = (const float*)d_in[4];
  const float* b1 = (const float*)d_in[5];
  const float* W2 = (const float*)d_in[6];
  const float* b2 = (const float*)d_in[7];
  const float* g1 = (const float*)d_in[8];
  const float* be1 = (const float*)d_in[9];
  const float* g2 = (const float*)d_in[10];
  const float* be2 = (const float*)d_in[11];
  float* out = (float*)d_out;

  const int M = 2 * SEQ; // 4096 rows
  char* ws = (char*)d_ws;
  size_t off = 0;
  auto carve = [&](size_t bytes) -> void* {
    void* p = (void*)(ws + off);
    off += (bytes + 255) & ~(size_t)255;
    return p;
  };
  const size_t headElems = (size_t)2 * HND * SEQ * 64;
  ushort_t* qb = (ushort_t*)carve(headElems * 2);
  ushort_t* kb = (ushort_t*)carve(headElems * 2);
  ushort_t* vtb = (ushort_t*)carve(headElems * 2);
  float* ctx = (float*)carve((size_t)M * DM * 4);
  float* res1 = (float*)carve((size_t)M * DM * 4);
  float* h = (float*)carve((size_t)M * DM * 4);
  float* ff = (float*)carve((size_t)M * 3072 * 4);
  float* res2 = (float*)carve((size_t)M * DM * 4);

  dim3 blk(128);
  // 1) QKV projection -> bf16 q (scaled), k, v-transposed head buffers
  gemm_kernel<1><<<dim3(2304 / 64, M / 128), blk, 0, stream>>>(
      x, Wqkv, nullptr, nullptr, nullptr, qb, kb, vtb, M, 2304, DM, 0, 0, 0);
  // 2) causal flash attention -> ctx [b][s][d]
  attn_kernel<<<dim3(SEQ / 64, 2 * HND), blk, 0, stream>>>(qb, kb, vtb, ctx);
  // 3) out projection + bout + residual(x)
  gemm_kernel<0><<<dim3(DM / 64, M / 128), blk, 0, stream>>>(
      ctx, Wout, bout, x, res1, nullptr, nullptr, nullptr, M, DM, DM, 1, 1, 0);
  // 4) LN1
  ln_kernel<<<dim3(M), dim3(256), 0, stream>>>(res1, g1, be1, h);
  // 5) FFN1 + b1 + relu
  gemm_kernel<0><<<dim3(3072 / 64, M / 128), blk, 0, stream>>>(
      h, W1, b1, nullptr, ff, nullptr, nullptr, nullptr, M, 3072, DM, 1, 0, 1);
  // 6) FFN2 + b2 + residual(h)
  gemm_kernel<0><<<dim3(DM / 64, M / 128), blk, 0, stream>>>(
      ff, W2, b2, h, res2, nullptr, nullptr, nullptr, M, DM, 3072, 1, 1, 0);
  // 7) LN2 -> out
  ln_kernel<<<dim3(M), dim3(256), 0, stream>>>(res2, g2, be2, out);
}
